// Network_44607530336718
// MI455X (gfx1250) — compile-verified, matrix operand order checked by reading, not run on hardware
//
#include <hip/hip_runtime.h>

typedef _Float16 v8h  __attribute__((ext_vector_type(8)));
typedef _Float16 v16h __attribute__((ext_vector_type(16)));
typedef float    v8f  __attribute__((ext_vector_type(8)));

#define TBITS     19
#define TSIZE     (1u << TBITS)
#define WIDTH     128
#define BLOCK_PTS 256   // 8 waves * 32 rows per wave
#define THREADS   256   // 8 wave32 per block

// Load a v16h WMMA fragment: two contiguous 16B runs at p and p+16 halves.
// Matches CDNA5 16-bit A/B layout: VGPRs 0-3 = first 8 halves, VGPRs 4-7 = next 8.
static __device__ __forceinline__ v16h load_frag(const _Float16* p) {
  v8h lo = *(const v8h*)(p);
  v8h hi = *(const v8h*)(p + 16);
  v16h r;
#pragma unroll
  for (int i = 0; i < 8; ++i) { r[i] = lo[i]; r[i + 8] = hi[i]; }
  return r;
}

// One MLP layer for this wave's 32 rows (two 16-row M-tiles sharing every B
// fragment -> half the LDS B-traffic per WMMA). act layout: [row][k] f16,
// stride WIDTH. Weights Ws layout: [n][k] f16, stride KDIM. bias f32. ReLU.
template <int KDIM>
static __device__ __forceinline__ void mlp_layer(_Float16* act, const _Float16* Ws,
                                                 const float* bias, int rowBase,
                                                 int col, int hi) {
  constexpr int NKT = KDIM / 32;
  const _Float16* ar0 = act + (size_t)(rowBase + col) * WIDTH + hi * 8;
  const _Float16* ar1 = ar0 + 16 * WIDTH;
  v16h a0[NKT], a1[NKT];
#pragma unroll
  for (int kt = 0; kt < NKT; ++kt) {
    a0[kt] = load_frag(ar0 + kt * 32);
    a1[kt] = load_frag(ar1 + kt * 32);
  }

  v8f acc0[8], acc1[8];
#pragma unroll
  for (int nt = 0; nt < 8; ++nt) {
    const int n = nt * 16 + col;
    const float bv = bias[n];
#pragma unroll
    for (int r = 0; r < 8; ++r) { acc0[nt][r] = bv; acc1[nt][r] = bv; }
    const _Float16* bw = Ws + (size_t)n * KDIM + hi * 8;
#pragma unroll
    for (int kt = 0; kt < NKT; ++kt) {
      v16h bf = load_frag(bw + kt * 32);  // one LDS B fragment ...
      acc0[nt] = __builtin_amdgcn_wmma_f32_16x16x32_f16(
          false, a0[kt], false, bf, (short)0, acc0[nt], false, false);
      acc1[nt] = __builtin_amdgcn_wmma_f32_16x16x32_f16(  // ... feeds two WMMAs
          false, a1[kt], false, bf, (short)0, acc1[nt], false, false);
    }
  }
  // ReLU + writeback. D layout: lane<16 VGPR r -> (M=r, N=lane);
  // lane>=16 VGPR r -> (M=8+r, N=lane-16).
#pragma unroll
  for (int nt = 0; nt < 8; ++nt) {
    const int n = nt * 16 + col;
#pragma unroll
    for (int r = 0; r < 8; ++r) {
      float v0 = acc0[nt][r];
      float v1 = acc1[nt][r];
      v0 = v0 > 0.f ? v0 : 0.f;
      v1 = v1 > 0.f ? v1 : 0.f;
      const int m = rowBase + hi * 8 + r;
      act[(size_t)m * WIDTH + n] = (_Float16)v0;
      act[(size_t)(m + 16) * WIDTH + n] = (_Float16)v1;
    }
  }
}

__global__ __launch_bounds__(THREADS)
void ngp_fused(const float* __restrict__ pos, const float* __restrict__ bmin,
               const float* __restrict__ bmax, const float* __restrict__ tables,
               const float* __restrict__ W0, const float* __restrict__ b0,
               const float* __restrict__ W1, const float* __restrict__ b1,
               const float* __restrict__ W2, const float* __restrict__ b2,
               const float* __restrict__ W3, const float* __restrict__ b3,
               const float* __restrict__ Wo, const float* __restrict__ bo,
               float* __restrict__ out, int npts) {
  extern __shared__ __align__(16) char smem[];
  _Float16* act = (_Float16*)smem;          // [256][128] activations, f16
  _Float16* w0s = act + 256 * 128;          // [128 n][32 k]
  _Float16* w1s = w0s + 128 * 32;           // [128 n][128 k]
  _Float16* w2s = w1s + 128 * 128;
  _Float16* w3s = w2s + 128 * 128;
  _Float16* wos = w3s + 128 * 128;          // [16 n][128 k] (n>=4 zero-padded)
  float* b0s = (float*)(wos + 16 * 128);
  float* b1s = b0s + 128;
  float* b2s = b1s + 128;
  float* b3s = b2s + 128;
  float* bos = b3s + 128;                   // [16]

  const int tid = threadIdx.x;
  const int lane = tid & 31;
  const int wv = tid >> 5;
  const int col = lane & 15;   // N column within tile
  const int hi = lane >> 4;    // K-half selector
  const int rowBase = wv * 32; // this wave owns rows [rowBase, rowBase+32)

  // ---- cooperative weight staging: global f32 -> LDS f16, transposed [n][k] ----
  for (int i = tid; i < 32 * 128; i += THREADS) {          // W0 (k,n) row-major in
    int k = i >> 7, n = i & 127;
    w0s[n * 32 + k] = (_Float16)W0[i];
  }
  for (int i = tid; i < 128 * 128; i += THREADS) {
    int k = i >> 7, n = i & 127;
    w1s[n * 128 + k] = (_Float16)W1[i];
  }
  for (int i = tid; i < 128 * 128; i += THREADS) {
    int k = i >> 7, n = i & 127;
    w2s[n * 128 + k] = (_Float16)W2[i];
  }
  for (int i = tid; i < 128 * 128; i += THREADS) {
    int k = i >> 7, n = i & 127;
    w3s[n * 128 + k] = (_Float16)W3[i];
  }
  for (int i = tid; i < 16 * 128; i += THREADS) {          // Wout padded to 16 cols
    int n = i >> 7, k = i & 127;
    wos[i] = (n < 4) ? (_Float16)Wo[k * 4 + n] : (_Float16)0.f;
  }
  for (int i = tid; i < 128; i += THREADS) {
    b0s[i] = b0[i]; b1s[i] = b1[i]; b2s[i] = b2[i]; b3s[i] = b3[i];
  }
  if (tid < 16) bos[tid] = (tid < 4) ? bo[tid] : 0.f;

  // ---- hash-grid encode: one lane per point, all 16 levels ----
  {
    const int row = rowBase + lane;
    const int g = blockIdx.x * BLOCK_PTS + row;
    // floor(16 * 2^(lv/3)) for lv = 0..15 (compile-time constants)
    constexpr float RESARR[16] = {16.f,  20.f,  25.f,  32.f,  40.f,  50.f,
                                  64.f,  80.f,  101.f, 128.f, 161.f, 203.f,
                                  256.f, 322.f, 406.f, 512.f};
    if (g < npts) {
      const float mnx = bmin[0], mny = bmin[1], mnz = bmin[2];
      const float xx = (pos[3 * g + 0] - mnx) / (bmax[0] - mnx);
      const float yy = (pos[3 * g + 1] - mny) / (bmax[1] - mny);
      const float zz = (pos[3 * g + 2] - mnz) / (bmax[2] - mnz);
#pragma unroll
      for (int lv = 0; lv < 16; ++lv) {
        const float r = RESARR[lv];
        const float sx = xx * r, sy = yy * r, sz = zz * r;
        const float fx = floorf(sx), fy = floorf(sy), fz = floorf(sz);
        const float tx = sx - fx, ty = sy - fy, tz = sz - fz;
        const unsigned ux0 = (unsigned)(int)fx;
        const unsigned uy0 = (unsigned)(int)fy;
        const unsigned uz0 = (unsigned)(int)fz;
        const float* tb = tables + ((size_t)lv << (TBITS + 1));  // lv*T*2 floats
        float a0 = 0.f, a1 = 0.f;
#pragma unroll
        for (int c = 0; c < 8; ++c) {
          const unsigned ux = ux0 + (c & 1);
          const unsigned uy = uy0 + ((c >> 1) & 1);
          const unsigned uz = uz0 + ((c >> 2) & 1);
          const unsigned h = ux ^ (uy * 2654435761u) ^ (uz * 805459861u);
          const unsigned idx = h & (TSIZE - 1u);
          const float wx = (c & 1) ? tx : 1.f - tx;
          const float wy = ((c >> 1) & 1) ? ty : 1.f - ty;
          const float wz = ((c >> 2) & 1) ? tz : 1.f - tz;
          const float w = wx * wy * wz;
          const float* fp = tb + ((size_t)idx * 2);  // L2-resident gather (64MB < 192MB)
          a0 += w * fp[0];
          a1 += w * fp[1];
        }
        act[(size_t)row * WIDTH + 2 * lv + 0] = (_Float16)a0;
        act[(size_t)row * WIDTH + 2 * lv + 1] = (_Float16)a1;
      }
    } else {
#pragma unroll
      for (int lv = 0; lv < 16; ++lv) {
        act[(size_t)row * WIDTH + 2 * lv + 0] = (_Float16)0.f;
        act[(size_t)row * WIDTH + 2 * lv + 1] = (_Float16)0.f;
      }
    }
  }
  __syncthreads();  // weights visible to all waves; act rows are wave-private

  // ---- fused MLP: each wave computes its own 32x128 tile, no further barriers ----
  mlp_layer<32>(act, w0s, b0s, rowBase, col, hi);
  mlp_layer<128>(act, w1s, b1s, rowBase, col, hi);
  mlp_layer<128>(act, w2s, b2s, rowBase, col, hi);
  mlp_layer<128>(act, w3s, b3s, rowBase, col, hi);

  // ---- output layer: 32x128 @ 128x16(pad of 4) -> 32x4, no ReLU ----
  {
    const _Float16* ar0 = act + (size_t)(rowBase + col) * WIDTH + hi * 8;
    const _Float16* ar1 = ar0 + 16 * WIDTH;
    v16h a0[4], a1[4];
#pragma unroll
    for (int kt = 0; kt < 4; ++kt) {
      a0[kt] = load_frag(ar0 + kt * 32);
      a1[kt] = load_frag(ar1 + kt * 32);
    }
    v8f acc0, acc1;
    const float bv = bos[col];
#pragma unroll
    for (int r = 0; r < 8; ++r) { acc0[r] = bv; acc1[r] = bv; }
    const _Float16* bw = wos + (size_t)col * WIDTH + hi * 8;
#pragma unroll
    for (int kt = 0; kt < 4; ++kt) {
      v16h bf = load_frag(bw + kt * 32);
      acc0 = __builtin_amdgcn_wmma_f32_16x16x32_f16(
          false, a0[kt], false, bf, (short)0, acc0, false, false);
      acc1 = __builtin_amdgcn_wmma_f32_16x16x32_f16(
          false, a1[kt], false, bf, (short)0, acc1, false, false);
    }
    if (col < 4) {
#pragma unroll
      for (int r = 0; r < 8; ++r) {
        const int m = rowBase + hi * 8 + r;
        const int g0 = blockIdx.x * BLOCK_PTS + m;
        const int g1 = g0 + 16;
        if (g0 < npts) out[(size_t)g0 * 4 + col] = acc0[r];
        if (g1 < npts) out[(size_t)g1 * 4 + col] = acc1[r];
      }
    }
  }
}

extern "C" void kernel_launch(void* const* d_in, const int* in_sizes, int n_in,
                              void* d_out, int out_size, void* d_ws, size_t ws_size,
                              hipStream_t stream) {
  const float* pos    = (const float*)d_in[0];
  const float* bmin   = (const float*)d_in[1];
  const float* bmax   = (const float*)d_in[2];
  const float* tables = (const float*)d_in[3];
  const float* W0 = (const float*)d_in[4];
  const float* b0 = (const float*)d_in[5];
  const float* W1 = (const float*)d_in[6];
  const float* b1 = (const float*)d_in[7];
  const float* W2 = (const float*)d_in[8];
  const float* b2 = (const float*)d_in[9];
  const float* W3 = (const float*)d_in[10];
  const float* b3 = (const float*)d_in[11];
  const float* Wo = (const float*)d_in[12];
  const float* bo = (const float*)d_in[13];

  const int npts = in_sizes[0] / 3;
  const int blocks = (npts + BLOCK_PTS - 1) / BLOCK_PTS;
  const size_t smem =
      (size_t)(256 * 128 + 128 * 32 + 3 * 128 * 128 + 16 * 128) * sizeof(_Float16) +
      (size_t)(4 * 128 + 16) * sizeof(float);

  ngp_fused<<<blocks, THREADS, smem, stream>>>(pos, bmin, bmax, tables,
                                               W0, b0, W1, b1, W2, b2, W3, b3,
                                               Wo, bo, (float*)d_out, npts);
}
